// StorageDFLAgent_15333033246829
// MI455X (gfx1250) — matrix operand order repaired and checked by reading, not compiled, not run on hardware
//
#include <hip/hip_runtime.h>

typedef _Float16 f16;
typedef __attribute__((ext_vector_type(16))) _Float16 v16h;
typedef __attribute__((ext_vector_type(8)))  _Float16 v8h;
typedef __attribute__((ext_vector_type(8)))  float    v8f;

#define BATCH_N   8192
#define STATE_D   512
#define HIDDEN_D  1024
#define HORIZON_N 24

#define WMMA_F16(a, b, c) \
  __builtin_amdgcn_wmma_f32_16x16x32_f16(false, (a), false, (b), (short)0, (c), false, false)

// ---------------------------------------------------------------- conversions
__global__ void cvt_f32_to_f16(const float* __restrict__ in, f16* __restrict__ out, int n) {
  int i = blockIdx.x * blockDim.x + threadIdx.x;
  if (i < n) out[i] = (f16)in[i];
}

// ---------------------------------------------------------------- WMMA frags
// 16-bit A 16x32 (MxK): lane half=0 holds K {0..7,16..23}, half=1 holds {8..15,24..31}
__device__ __forceinline__ v16h load_a_frag(const f16* __restrict__ row, int k0, int half) {
  v8h lo = *reinterpret_cast<const v8h*>(row + k0 + 8 * half);
  v8h hi = *reinterpret_cast<const v8h*>(row + k0 + 16 + 8 * half);
  return __builtin_shufflevector(lo, hi, 0,1,2,3,4,5,6,7,8,9,10,11,12,13,14,15);
}
// 16-bit B 32x16 (KxN): lane = column N; half=0 holds K {0..7,16..23}, half=1 holds {8..15,24..31}
__device__ __forceinline__ v16h load_b_frag(const f16* __restrict__ row, int k0, int half) {
  v8h lo = *reinterpret_cast<const v8h*>(row + k0 + 8 * half);
  v8h hi = *reinterpret_cast<const v8h*>(row + k0 + 16 + 8 * half);
  return __builtin_shufflevector(lo, hi, 0,1,2,3,4,5,6,7,8,9,10,11,12,13,14,15);
}

template <bool RELU, bool OUT_F16>
__device__ __forceinline__ void store_tile(const v8f& c, int mT, int nT, int half, int l16,
                                           float bv, float* __restrict__ Cf,
                                           f16* __restrict__ Ch, int N) {
#pragma unroll
  for (int r = 0; r < 8; ++r) {
    int row = mT + r + 8 * half;       // C/D layout: VGPR r -> M = r + 8*half
    int col = nT + l16;                // lane%16 -> N
    float v = c[r] + bv;
    if (RELU) v = fmaxf(v, 0.0f);
    if (OUT_F16) Ch[(size_t)row * N + col] = (f16)v;
    else         Cf[(size_t)row * N + col] = v;
  }
}

// GEMM: C[M,N] = act(A[M,K] * W[N,K]^T + bias).  Block = 256 thr = 8 waves,
// block tile 128(M) x 64(N), wave tile 32x32 (2x2 WMMA tiles), K step 32.
template <bool RELU, bool OUT_F16>
__global__ __launch_bounds__(256) void gemm_wmma(const f16* __restrict__ A,
                                                 const f16* __restrict__ W,
                                                 const float* __restrict__ bias,
                                                 float* __restrict__ Cf,
                                                 f16* __restrict__ Ch,
                                                 int M, int N, int K) {
  const int waveId = threadIdx.x >> 5;
  const int lane   = threadIdx.x & 31;
  const int half   = lane >> 4;
  const int l16    = lane & 15;
  const int waveM  = waveId & 3;
  const int waveN  = waveId >> 2;
  const int mBase  = blockIdx.x * 128 + waveM * 32;
  const int nBase  = blockIdx.y * 64  + waveN * 32;

  const f16* arow0 = A + (size_t)(mBase + l16)      * K;
  const f16* arow1 = A + (size_t)(mBase + 16 + l16) * K;
  const f16* brow0 = W + (size_t)(nBase + l16)      * K;
  const f16* brow1 = W + (size_t)(nBase + 16 + l16) * K;

  v8f c00 = {}, c01 = {}, c10 = {}, c11 = {};

  for (int k0 = 0; k0 < K; k0 += 32) {
    __builtin_prefetch(arow0 + k0 + 128, 0, 1);   // gfx1250 global_prefetch_b8 path
    __builtin_prefetch(brow0 + k0 + 128, 0, 1);
    v16h a0 = load_a_frag(arow0, k0, half);
    v16h a1 = load_a_frag(arow1, k0, half);
    v16h b0 = load_b_frag(brow0, k0, half);
    v16h b1 = load_b_frag(brow1, k0, half);
    c00 = WMMA_F16(a0, b0, c00);
    c01 = WMMA_F16(a0, b1, c01);
    c10 = WMMA_F16(a1, b0, c10);
    c11 = WMMA_F16(a1, b1, c11);
  }

  const float bv0 = bias[nBase + l16];
  const float bv1 = bias[nBase + 16 + l16];
  store_tile<RELU, OUT_F16>(c00, mBase,      nBase,      half, l16, bv0, Cf, Ch, N);
  store_tile<RELU, OUT_F16>(c01, mBase,      nBase + 16, half, l16, bv1, Cf, Ch, N);
  store_tile<RELU, OUT_F16>(c10, mBase + 16, nBase,      half, l16, bv0, Cf, Ch, N);
  store_tile<RELU, OUT_F16>(c11, mBase + 16, nBase + 16, half, l16, bv1, Cf, Ch, N);
}

// ------------------------------------------- LayerNorm + 24-wide projection
// One wave (32 lanes) per batch row; 32 elements per lane in registers.
__global__ __launch_bounds__(256) void ln_prices_kernel(const float* __restrict__ H2,
                                                        const float* __restrict__ ln_g,
                                                        const float* __restrict__ ln_b,
                                                        const float* __restrict__ Wp,
                                                        const float* __restrict__ bp,
                                                        float* __restrict__ prices) {
  const int wave = threadIdx.x >> 5;
  const int lane = threadIdx.x & 31;
  const int row  = blockIdx.x * (blockDim.x >> 5) + wave;
  const float* h = H2 + (size_t)row * HIDDEN_D;

  float v[32];
  float s = 0.0f, ss = 0.0f;
#pragma unroll
  for (int i = 0; i < 32; ++i) {
    float x = h[lane + 32 * i];
    v[i] = x; s += x; ss += x * x;
  }
#pragma unroll
  for (int m = 16; m >= 1; m >>= 1) {
    s  += __shfl_xor(s,  m, 32);
    ss += __shfl_xor(ss, m, 32);
  }
  const float mean = s * (1.0f / HIDDEN_D);
  const float var  = ss * (1.0f / HIDDEN_D) - mean * mean;
  const float rstd = rsqrtf(var + 1e-5f);
#pragma unroll
  for (int i = 0; i < 32; ++i) {
    int k = lane + 32 * i;
    v[i] = (v[i] - mean) * rstd * ln_g[k] + ln_b[k];
  }
  for (int hh = 0; hh < HORIZON_N; ++hh) {
    const float* w = Wp + (size_t)hh * HIDDEN_D;
    float acc = 0.0f;
#pragma unroll
    for (int i = 0; i < 32; ++i) acc += v[i] * w[lane + 32 * i];
#pragma unroll
    for (int m = 16; m >= 1; m >>= 1) acc += __shfl_xor(acc, m, 32);
    if (lane == 0) prices[(size_t)row * HORIZON_N + hh] = acc + bp[hh];
  }
}

// ---------------------------------------------------------------- LP solver
// Projected gradient ascent, 200 iters, one thread per sample, horizon in regs.
__global__ __launch_bounds__(256) void solver_kernel(const float* __restrict__ prices,
                                                     const float* __restrict__ soc0v,
                                                     const float* __restrict__ degv,
                                                     float* __restrict__ charge,
                                                     float* __restrict__ discharge) {
  const int i = blockIdx.x * blockDim.x + threadIdx.x;
  if (i >= BATCH_N) return;

  const float ETA = 0.95f, INV_ETA = 1.0f / 0.95f;
  const float SMIN = 0.5f, SMAX = 9.5f, PMAX = 2.0f;
  const float MU2 = 100.0f;           // 2*MU
  const float EPS_REG = 1e-3f, LR = 0.05f;

  float p[HORIZON_N];
#pragma unroll
  for (int t = 0; t < HORIZON_N; ++t) p[t] = prices[(size_t)i * HORIZON_N + t];
  const float soc0 = soc0v[i];
  const float deg  = degv[i];

  float c[HORIZON_N], d[HORIZON_N];
#pragma unroll
  for (int t = 0; t < HORIZON_N; ++t) { c[t] = 0.0f; d[t] = 0.0f; }

  for (int it = 0; it < 200; ++it) {
    // forward scan: soc_t and dPen/dsoc_t
    float g[HORIZON_N];
    float soc = soc0;
#pragma unroll
    for (int t = 0; t < HORIZON_N; ++t) {
      soc += ETA * c[t] - d[t] * INV_ETA;
      float over  = soc - SMAX;
      float under = SMIN - soc;
      float gg = 0.0f;
      if (over  > 0.0f) gg += MU2 * over;
      if (under > 0.0f) gg -= MU2 * under;
      g[t] = gg;
    }
    // reverse scan: suffix sum S_k = sum_{t>=k} g_t, simultaneous update + clip
    float S = 0.0f;
#pragma unroll
    for (int t = HORIZON_N - 1; t >= 0; --t) {
      S += g[t];
      float gc = -p[t] - deg - EPS_REG * c[t] - ETA * S;
      float gd =  p[t] - deg - EPS_REG * d[t] + INV_ETA * S;
      c[t] = fminf(fmaxf(c[t] + LR * gc, 0.0f), PMAX);
      d[t] = fminf(fmaxf(d[t] + LR * gd, 0.0f), PMAX);
    }
  }

#pragma unroll
  for (int t = 0; t < HORIZON_N; ++t) {
    charge[(size_t)i * HORIZON_N + t]    = c[t];
    discharge[(size_t)i * HORIZON_N + t] = d[t];
  }
}

// ---------------------------------------------------------------- launcher
extern "C" void kernel_launch(void* const* d_in, const int* in_sizes, int n_in,
                              void* d_out, int out_size, void* d_ws, size_t ws_size,
                              hipStream_t stream) {
  const float* x    = (const float*)d_in[0];   // [8192,512]
  const float* soc0 = (const float*)d_in[1];   // [8192,1]
  const float* deg  = (const float*)d_in[2];   // [8192,1]
  const float* W1   = (const float*)d_in[3];   // [1024,512]
  const float* b1   = (const float*)d_in[4];   // [1024]
  const float* W2   = (const float*)d_in[5];   // [1024,1024]
  const float* b2   = (const float*)d_in[6];   // [1024]
  const float* ln_g = (const float*)d_in[7];   // [1024]
  const float* ln_b = (const float*)d_in[8];   // [1024]
  const float* Wp   = (const float*)d_in[9];   // [24,1024]
  const float* bp   = (const float*)d_in[10];  // [24]
  float* out = (float*)d_out;

  float* charge    = out;                                   // [8192,24]
  float* discharge = out + (size_t)BATCH_N * HORIZON_N;     // [8192,24]
  float* prices    = out + (size_t)2 * BATCH_N * HORIZON_N; // [8192,24]

  // workspace layout (f16 activations/weights, f32 pre-LN hidden)
  char* ws = (char*)d_ws;
  f16*   Xh  = (f16*)(ws);                                   //  8 MiB
  f16*   W1h = (f16*)(ws + (size_t)8  * 1024 * 1024);        //  1 MiB
  f16*   W2h = (f16*)(ws + (size_t)9  * 1024 * 1024);        //  2 MiB
  f16*   H1h = (f16*)(ws + (size_t)11 * 1024 * 1024);        // 16 MiB
  float* H2  = (float*)(ws + (size_t)27 * 1024 * 1024);      // 32 MiB
  (void)ws_size; (void)n_in; (void)in_sizes; (void)out_size;

  // 1) convert inputs to f16
  {
    int n = BATCH_N * STATE_D;
    cvt_f32_to_f16<<<(n + 255) / 256, 256, 0, stream>>>(x, Xh, n);
    n = HIDDEN_D * STATE_D;
    cvt_f32_to_f16<<<(n + 255) / 256, 256, 0, stream>>>(W1, W1h, n);
    n = HIDDEN_D * HIDDEN_D;
    cvt_f32_to_f16<<<(n + 255) / 256, 256, 0, stream>>>(W2, W2h, n);
  }

  // 2) H1 = relu(x @ W1^T + b1)   [8192,1024] f16
  {
    dim3 grid(BATCH_N / 128, HIDDEN_D / 64);
    gemm_wmma<true, true><<<grid, 256, 0, stream>>>(Xh, W1h, b1, nullptr, H1h,
                                                    BATCH_N, HIDDEN_D, STATE_D);
  }
  // 3) H2 = H1 @ W2^T + b2        [8192,1024] f32
  {
    dim3 grid(BATCH_N / 128, HIDDEN_D / 64);
    gemm_wmma<false, false><<<grid, 256, 0, stream>>>(H1h, W2h, b2, H2, nullptr,
                                                      BATCH_N, HIDDEN_D, HIDDEN_D);
  }
  // 4) prices = LN(H2)*g+b @ Wp^T + bp   -> d_out[2*B*H ..]
  ln_prices_kernel<<<BATCH_N / 8, 256, 0, stream>>>(H2, ln_g, ln_b, Wp, bp, prices);

  // 5) 200-iteration projected gradient ascent solver -> charge/discharge
  solver_kernel<<<BATCH_N / 256, 256, 0, stream>>>(prices, soc0, deg, charge, discharge);
}
